// GCN_Net2_48206712930320
// MI455X (gfx1250) — matrix operand order repaired and checked by reading, not compiled, not run on hardware
//
#include <hip/hip_runtime.h>
#include <hip/hip_bf16.h>
#include <math.h>

// ---------------------------------------------------------------------------
// GCN (2-layer) for MI455X / gfx1250.
//  - Dense transforms: wave32 WMMA f32 <- f16 x f16 (16x16x32), f32 accum.
//    Weights pre-packed to f16 column-major (zero-padded) so the inner loop
//    is branch-free vector loads.
//  - Edge aggregation: L2-resident gather + global_atomic_add_f32 scatter
//    (entire working set ~77 MB << 192 MB L2).
// ---------------------------------------------------------------------------

typedef __attribute__((ext_vector_type(16))) _Float16 v16h;
typedef __attribute__((ext_vector_type(8)))  float    v8f;

// ---------------- degree / normalization ----------------

__global__ void deg_init_kernel(float* deg, int n) {
    int i = blockIdx.x * blockDim.x + threadIdx.x;
    if (i < n) deg[i] = 1.0f;   // self-loop contributes 1
}

__global__ void deg_count_kernel(const int* __restrict__ dst, float* deg, int e) {
    int i = blockIdx.x * blockDim.x + threadIdx.x;
    if (i < e) atomicAdd(&deg[dst[i]], 1.0f);
}

__global__ void deg_to_dinv_kernel(float* deg, int n) {
    int i = blockIdx.x * blockDim.x + threadIdx.x;
    if (i < n) deg[i] = rsqrtf(deg[i]);  // deg >= 1 always (self-loop)
}

// ---------------- weight pre-pack ----------------
// B[K x N] fp32 row-major  ->  Bt[Npad x K] f16 column-major, zero padded.
__global__ void pack_b_kernel(const float* __restrict__ B,
                              _Float16* __restrict__ Bt,
                              int K, int N, int Npad) {
    unsigned t = blockIdx.x * blockDim.x + threadIdx.x;
    unsigned total = (unsigned)Npad * (unsigned)K;
    if (t >= total) return;
    unsigned col = t / (unsigned)K;
    unsigned k   = t - col * (unsigned)K;
    float v = (col < (unsigned)N) ? B[(size_t)k * N + col] : 0.0f;
    Bt[t] = (_Float16)v;
}

// ---------------- WMMA GEMM: C[M x N] = A[M x K] * B[K x N] ----------------
// A row-major fp32 (converted to f16 in-register); Bt pre-packed f16
// column-major [Npad x K]. One wave per 16x16 C tile; blockDim = (32, 4).
// K must be a multiple of 32; M a multiple of 16 handled by uniform guard.

__global__ void wmma_gemm_kernel(const float* __restrict__ A,
                                 const _Float16* __restrict__ Bt,
                                 float* __restrict__ C,
                                 int M, int N, int K) {
    const int lane  = threadIdx.x;          // 0..31
    const int mtile = blockIdx.y * blockDim.y + threadIdx.y;
    const int ntile = blockIdx.x;
    if (mtile * 16 >= M) return;

    const int row = lane & 15;   // A row / B col / C col within tile
    const int hi  = lane >> 4;   // lane half selects K sub-range

    const int m0  = mtile * 16;
    const int n0  = ntile * 16;
    const int col = n0 + row;

    const float*    arow = A + (size_t)(m0 + row) * K;
    const _Float16* bcol = Bt + (size_t)col * K;

    v8f acc = {};

    for (int k0 = 0; k0 < K; k0 += 32) {
        // ---- A tile 16x32 (ISA 7.12.2 16-bit A layout) ----
        // lane half `hi`: VGPR0..3 hold K = k0+hi*8+{0..7} (pairs),
        //                 VGPR4..7 hold K = k0+16+hi*8+{0..7}.
        const float* ap = arow + k0 + hi * 8;
        float4 x0 = *(const float4*)(ap + 0);
        float4 x1 = *(const float4*)(ap + 4);
        float4 x2 = *(const float4*)(ap + 16);
        float4 x3 = *(const float4*)(ap + 20);
        v16h a;
        a[0]  = (_Float16)x0.x;  a[1]  = (_Float16)x0.y;
        a[2]  = (_Float16)x0.z;  a[3]  = (_Float16)x0.w;
        a[4]  = (_Float16)x1.x;  a[5]  = (_Float16)x1.y;
        a[6]  = (_Float16)x1.z;  a[7]  = (_Float16)x1.w;
        a[8]  = (_Float16)x2.x;  a[9]  = (_Float16)x2.y;
        a[10] = (_Float16)x2.z;  a[11] = (_Float16)x2.w;
        a[12] = (_Float16)x3.x;  a[13] = (_Float16)x3.y;
        a[14] = (_Float16)x3.z;  a[15] = (_Float16)x3.w;

        // ---- B tile 32x16: lane half `hi`: elems 0..7 K=k0+hi*8+{0..7},
        // elems 8..15 K=k0+16+hi*8+{0..7}; col = n0 + (lane&15). Column-major f16.
        typedef __attribute__((ext_vector_type(8))) _Float16 v8h_t;
        v8h_t b0 = *(const v8h_t*)(bcol + k0 + hi * 8);
        v8h_t b1 = *(const v8h_t*)(bcol + k0 + 16 + hi * 8);
        v16h b = __builtin_shufflevector(b0, b1, 0, 1, 2, 3, 4, 5, 6, 7, 8, 9, 10, 11, 12, 13, 14, 15);

        acc = __builtin_amdgcn_wmma_f32_16x16x32_f16(
            /*neg_a=*/false, a, /*neg_b=*/false, b,
            /*c_mod=*/(short)0, acc, /*reuse_a=*/false, /*reuse_b=*/false);
    }

    // ---- store: lanes 0-15: M=v, N=lane; lanes 16-31: M=v+8, N=lane-16 ----
    if (col < N) {
        #pragma unroll
        for (int v = 0; v < 8; ++v) {
            int m = m0 + v + 8 * hi;
            C[(size_t)m * N + col] = acc[v];
        }
    }
}

// ---------------- aggregation ----------------

// agg[n, f] = H[n, f] * dinv[n]^2 + bias[f]   (self-loop + bias; fully
// initializes the poisoned destination buffer)
__global__ void init_agg_kernel(const float* __restrict__ H,
                                const float* __restrict__ dinv,
                                const float* __restrict__ bias,
                                float* __restrict__ agg,
                                int n, int F) {
    unsigned t = blockIdx.x * blockDim.x + threadIdx.x;
    unsigned total = (unsigned)n * (unsigned)F;
    if (t >= total) return;
    unsigned node = t / (unsigned)F;
    unsigned f    = t - node * (unsigned)F;
    float w = dinv[node];
    agg[t] = H[t] * w * w + bias[f];
}

// agg[dst, f..f+3] += H[src, f..f+3] * dinv[src]*dinv[dst]
__global__ void edge_scatter_kernel(const int* __restrict__ src,
                                    const int* __restrict__ dst,
                                    const float* __restrict__ dinv,
                                    const float* __restrict__ H,
                                    float* __restrict__ agg,
                                    int e, int F) {
    unsigned t = blockIdx.x * blockDim.x + threadIdx.x;
    unsigned perEdge = (unsigned)(F >> 2);          // F is a multiple of 4
    unsigned total = (unsigned)e * perEdge;
    if (t >= total) return;
    unsigned edge = t / perEdge;
    unsigned f    = (t - edge * perEdge) << 2;

    int s = src[edge];
    int d = dst[edge];
    float w = dinv[s] * dinv[d];

    const float4 h = *(const float4*)(H + (size_t)s * F + f);
    float* out = agg + (size_t)d * F + f;
    atomicAdd(out + 0, h.x * w);
    atomicAdd(out + 1, h.y * w);
    atomicAdd(out + 2, h.z * w);
    atomicAdd(out + 3, h.w * w);
}

__global__ void relu_kernel(float* x, unsigned total) {
    unsigned t = blockIdx.x * blockDim.x + threadIdx.x;
    if (t < total) x[t] = fmaxf(x[t], 0.0f);
}

// ---------------- log_softmax (one wave32 per row) ----------------

__global__ void log_softmax_kernel(float* __restrict__ x, int n, int C) {
    int wavesPerBlock = blockDim.x >> 5;
    int node = blockIdx.x * wavesPerBlock + ((int)threadIdx.x >> 5);
    int lane = threadIdx.x & 31;
    if (node >= n) return;

    float* row = x + (size_t)node * C;
    float v0 = (lane      < C) ? row[lane]      : -__builtin_inff();
    float v1 = (lane + 32 < C) ? row[lane + 32] : -__builtin_inff();

    float m = fmaxf(v0, v1);
    #pragma unroll
    for (int off = 16; off > 0; off >>= 1)
        m = fmaxf(m, __shfl_xor(m, off, 32));

    float s = 0.0f;
    if (lane      < C) s += __expf(v0 - m);
    if (lane + 32 < C) s += __expf(v1 - m);
    #pragma unroll
    for (int off = 16; off > 0; off >>= 1)
        s += __shfl_xor(s, off, 32);

    float lse = m + __logf(s);
    if (lane      < C) row[lane]      = v0 - lse;
    if (lane + 32 < C) row[lane + 32] = v1 - lse;
}

// ---------------- host side ----------------

static inline unsigned cdiv(unsigned a, unsigned b) { return (a + b - 1) / b; }

extern "C" void kernel_launch(void* const* d_in, const int* in_sizes, int n_in,
                              void* d_out, int out_size, void* d_ws, size_t ws_size,
                              hipStream_t stream) {
    const float* X  = (const float*)d_in[0];   // [N, F_IN]
    const int*   ei = (const int*)d_in[1];     // [2, E]
    const float* W1 = (const float*)d_in[2];   // [F_IN, HID]
    const float* b1 = (const float*)d_in[3];   // [HID]
    const float* W2 = (const float*)d_in[4];   // [HID, NC]
    const float* b2 = (const float*)d_in[5];   // [NC]

    const int HID  = in_sizes[3];              // 128
    const int NC   = in_sizes[5];              // 40
    const int F_IN = in_sizes[2] / HID;        // 128
    const int N    = in_sizes[0] / F_IN;       // 50000
    const int E    = in_sizes[1] / 2;          // 800000

    const int NCpad = (unsigned)cdiv(NC, 16) * 16;   // 48

    const int* src = ei;
    const int* dst = ei + E;

    float* ws   = (float*)d_ws;
    float* dinv = ws;                               // N floats
    float* H    = dinv + N;                         // N*HID floats (H2 reuses N*NC)
    float* agg1 = H + (size_t)N * HID;              // N*HID floats
    _Float16* Bt1 = (_Float16*)(agg1 + (size_t)N * HID);   // HID * F_IN halves
    _Float16* Bt2 = Bt1 + (size_t)HID * F_IN;              // NCpad * HID halves
    float* out  = (float*)d_out;                    // N*NC floats (layer-2 agg in place)

    // 0) pre-pack weights to f16 column-major (zero padded)
    pack_b_kernel<<<cdiv((unsigned)HID * F_IN, 256), 256, 0, stream>>>(W1, Bt1, F_IN, HID, HID);
    pack_b_kernel<<<cdiv((unsigned)NCpad * HID, 256), 256, 0, stream>>>(W2, Bt2, HID, NC, NCpad);

    // 1) degree + symmetric norm
    deg_init_kernel<<<cdiv(N, 256), 256, 0, stream>>>(dinv, N);
    deg_count_kernel<<<cdiv(E, 256), 256, 0, stream>>>(dst, dinv, E);
    deg_to_dinv_kernel<<<cdiv(N, 256), 256, 0, stream>>>(dinv, N);

    // 2) H1 = X @ W1   (WMMA)
    {
        dim3 block(32, 4);
        dim3 grid(cdiv(HID, 16), cdiv(cdiv(N, 16), 4));
        wmma_gemm_kernel<<<grid, block, 0, stream>>>(X, Bt1, H, N, HID, F_IN);
    }

    // 3) agg1 = D^-1/2 (A+I) D^-1/2 H1 + b1 ; ReLU
    {
        unsigned total = (unsigned)N * (unsigned)HID;
        init_agg_kernel<<<cdiv(total, 256), 256, 0, stream>>>(H, dinv, b1, agg1, N, HID);
        unsigned work = (unsigned)E * (unsigned)(HID >> 2);
        edge_scatter_kernel<<<cdiv(work, 256), 256, 0, stream>>>(src, dst, dinv, H, agg1, E, HID);
        relu_kernel<<<cdiv(total, 256), 256, 0, stream>>>(agg1, total);
    }

    // 4) H2 = agg1 @ W2   (WMMA, N=40 -> padded B, guarded store; ld = NC)
    {
        dim3 block(32, 4);
        dim3 grid(cdiv(NCpad, 16), cdiv(cdiv(N, 16), 4));
        wmma_gemm_kernel<<<grid, block, 0, stream>>>(agg1, Bt2, H, N, NC, HID);
    }

    // 5) out = D^-1/2 (A+I) D^-1/2 H2 + b2 ; log_softmax
    {
        unsigned total = (unsigned)N * (unsigned)NC;
        init_agg_kernel<<<cdiv(total, 256), 256, 0, stream>>>(H, dinv, b2, out, N, NC);
        unsigned work = (unsigned)E * (unsigned)(NC >> 2);
        edge_scatter_kernel<<<cdiv(work, 256), 256, 0, stream>>>(src, dst, dinv, H, out, E, NC);
        int wavesPerBlock = 8;  // 256 threads
        log_softmax_kernel<<<cdiv(N, wavesPerBlock), wavesPerBlock * 32, 0, stream>>>(out, N, NC);
    }
}
